// RandomBinaryConvV1_11682311045439
// MI455X (gfx1250) — matrix operand order repaired and checked by reading, not compiled, not run on hardware
//
#include <hip/hip_runtime.h>
#include <stdint.h>

typedef __attribute__((ext_vector_type(16))) _Float16 v16h;
typedef __attribute__((ext_vector_type(8)))  _Float16 v8h;
typedef __attribute__((ext_vector_type(2)))  _Float16 v2h;
typedef __attribute__((ext_vector_type(8)))  float    v8f;
typedef __attribute__((ext_vector_type(4)))  unsigned int v4u;
typedef __attribute__((ext_vector_type(8)))  int      v8i;
typedef __attribute__((ext_vector_type(4)))  int      v4i;

#define B_  16
#define C_  64
#define H_  128
#define W_  128
#define OC_ 128
#define HP  (H_ + 2)
#define WP  (W_ + 2)
#define LDSROW 72                     // halves per LDS row: 64 data + 8 pad (bank-conflict-free)
#define LDS_BYTES (9 * OC_ * LDSROW * 2)   // 165,888 B  (fits 320KB WGP LDS)

// ---------------------------------------------------------------------------
// Kernel 1: NCHW f32 -> padded NHWC f16, channels swizzled per 32-block as
// {0-7,16-23,8-15,24-31} so a WMMA A-fragment (16x32 f16 layout: lanes0-15
// hold K 0-7 & 16-23, lanes16-31 hold K 8-15 & 24-31) is one contiguous
// 32B load per lane at offset kk*32 + half*16.
// ---------------------------------------------------------------------------
__global__ __launch_bounds__(256) void pack_x_kernel(const float* __restrict__ x,
                                                     _Float16* __restrict__ wx) {
    const int wave = (blockIdx.x * blockDim.x + threadIdx.x) >> 5;
    const int lane = threadIdx.x & 31;
    if (wave >= B_ * HP) return;
    const int b  = wave / HP;
    const int hp = wave % HP;
    const int h  = hp - 1;
    const bool hin = (h >= 0) && (h < H_);

    const int q  = lane * 2;
    const int qq = q & 31, blk = q >> 5;
    const int c0 = blk * 32 + (qq < 8 ? qq : qq < 16 ? qq + 8 : qq < 24 ? qq - 8 : qq);

    const int hc = hin ? h : 0;
    const float* xr0 = x + (((size_t)(b * C_ + c0    ) * H_ + hc) * W_);
    const float* xr1 = x + (((size_t)(b * C_ + c0 + 1) * H_ + hc) * W_);
    _Float16* orow = wx + ((size_t)(b * HP + hp) * WP) * C_ + q;

    for (int wp = 0; wp < WP; ++wp) {
        const int w = wp - 1;
        v2h v = {(_Float16)0.f, (_Float16)0.f};
        if (hin && (unsigned)w < (unsigned)W_) {
            v.x = (_Float16)xr0[w];
            v.y = (_Float16)xr1[w];
        }
        *(v2h*)(orow + (size_t)wp * C_) = v;
    }
}

// ---------------------------------------------------------------------------
// Kernel 2: weight (CK=576, OC=128) f32 -> ww[r][oc][c] f16  (r = kh*3+kw).
// ---------------------------------------------------------------------------
__global__ __launch_bounds__(256) void pack_w_kernel(const float* __restrict__ wsrc,
                                                     _Float16* __restrict__ ww) {
    const int t = blockIdx.x * blockDim.x + threadIdx.x;
    if (t >= 9 * OC_ * C_) return;
    const int c    = t & (C_ - 1);
    const int rest = t >> 6;
    const int oc   = rest & (OC_ - 1);
    const int r    = rest >> 7;
    ww[t] = (_Float16)wsrc[(size_t)(c * 9 + r) * OC_ + oc];
}

// ---------------------------------------------------------------------------
// Kernel 3: implicit GEMM conv.
//  * Whole packed weight (147,456 B) staged LDS-resident via one TDM
//    tensor_load_to_lds per block, with D# padding (4 dwords per 32 dwords)
//    -> LDS row stride 144B, bank-conflict-free ds_load_b128 B-fragments.
//  * Wave = TWO 16-pixel M-tiles (32 pixels) x all 8 OC tiles:
//    per k-step 2 A loads + 8 B frags feed 16 WMMAs -> 288 wmma/wave.
// ---------------------------------------------------------------------------
__global__ __launch_bounds__(256) void conv_wmma_kernel(const _Float16* __restrict__ wx,
                                                        const _Float16* __restrict__ ww,
                                                        float* __restrict__ out) {
    extern __shared__ _Float16 lds_w[];

    // ---- TDM stage: global ww -> LDS (one wave issues; TDM ignores EXEC) ----
    if ((threadIdx.x >> 5) == 0) {
        const uint64_t ga = (uint64_t)(uintptr_t)ww;
        const uint32_t lb = (uint32_t)(uintptr_t)lds_w;   // LDS aperture: low 32 bits = LDS offset
        // D# group0: count=1 | lds_addr | global_addr[56:0] | type=2
        v4u g0 = { 1u, lb, (uint32_t)ga,
                   (uint32_t)((ga >> 32) & 0x01FFFFFFu) | 0x80000000u };
        // D# group1:
        //  w0: data_size=2 (4B) | pad_enable | pad_interval=4 (32dw) | pad_amount=3 (4dw)
        //  w1[31:16]=tensor_dim0.lo=36864(0x9000); w2[31:16]=tensor_dim1=1
        //  w3[31:16]=tile_dim0=0x9000; w4[15:0]=tile_dim1=1; w5=tensor_dim0_stride=36864
        v8i g1 = { (int)0x07120000, (int)0x90000000, (int)0x00010000, (int)0x90000000,
                   1, 36864, 0, 0 };
        v4i gz4 = { 0, 0, 0, 0 };
        v8i gz8 = { 0, 0, 0, 0, 0, 0, 0, 0 };
        __builtin_amdgcn_tensor_load_to_lds(g0, g1, gz4, gz4, gz8, 0);
        __builtin_amdgcn_s_wait_tensorcnt(0);
    }
    __syncthreads();

    const int lane = threadIdx.x & 31;
    const int m    = lane & 15;
    const int half = lane >> 4;

    const int mt = blockIdx.x * 8 + (threadIdx.x >> 5);
    const int wt = mt & 3;            // 4 double-tiles per row
    const int bh = mt >> 2;
    const int h  = bh & (H_ - 1);
    const int b  = bh >> 7;
    const int wbase = wt * 32;

    v8f acc0[8], acc1[8];
#pragma unroll
    for (int i = 0; i < 8; ++i) { v8f z = {}; acc0[i] = z; acc1[i] = z; }

#pragma unroll
    for (int r = 0; r < 9; ++r) {
        const int dh = r / 3 - 1, dw = r % 3 - 1;
        const _Float16* ap = wx
            + ((size_t)((b * HP + (h + 1 + dh)) * WP) + (wbase + 1 + m + dw)) * C_
            + half * 16;
        const int rowb = r * OC_;     // LDS row base (row = oc)
#pragma unroll
        for (int kk = 0; kk < 2; ++kk) {
            const v16h a0 = *(const v16h*)(ap + kk * 32);              // tile0 A frag
            const v16h a1 = *(const v16h*)(ap + 16 * C_ + kk * 32);    // tile1 A frag
#pragma unroll
            for (int nt = 0; nt < 8; ++nt) {
                const _Float16* bp = lds_w
                    + (size_t)(rowb + nt * 16 + m) * LDSROW + kk * 32 + half * 8;
                const v8h blo = *(const v8h*)(bp);
                const v8h bhi = *(const v8h*)(bp + 16);
                const v16h bm = __builtin_shufflevector(blo, bhi,
                    0,1,2,3,4,5,6,7,8,9,10,11,12,13,14,15);
                acc0[nt] = __builtin_amdgcn_wmma_f32_16x16x32_f16(
                    false, a0, false, bm, (short)0, acc0[nt], false, false);
                acc1[nt] = __builtin_amdgcn_wmma_f32_16x16x32_f16(
                    false, a1, false, bm, (short)0, acc1[nt], false, false);
            }
        }
    }

    // D layout: lanes0-15 M=v, lanes16-31 M=v+8; N = lane&15.
    float* op = out + (((size_t)(b * OC_) * H_ + h) * W_) + wbase + half * 8;
#pragma unroll
    for (int nt = 0; nt < 8; ++nt) {
        const int oc = nt * 16 + m;
        *(v8f*)(op + (size_t)oc * H_ * W_)      = acc0[nt];
        *(v8f*)(op + (size_t)oc * H_ * W_ + 16) = acc1[nt];
    }
}

// ---------------------------------------------------------------------------
extern "C" void kernel_launch(void* const* d_in, const int* in_sizes, int n_in,
                              void* d_out, int out_size, void* d_ws, size_t ws_size,
                              hipStream_t stream) {
    const float* x = (const float*)d_in[0];   // (16,64,128,128) f32
    const float* w = (const float*)d_in[1];   // (576,128) f32
    float* out = (float*)d_out;               // (16,128,128,128) f32

    _Float16* wx = (_Float16*)d_ws;                                   // 34.6 MB
    _Float16* ww = (_Float16*)((char*)d_ws +
                   (size_t)B_ * HP * WP * C_ * sizeof(_Float16));     // +147 KB

    static_assert(LDS_BYTES == 165888, "lds size");
    (void)hipFuncSetAttribute((const void*)conv_wmma_kernel,
                              hipFuncAttributeMaxDynamicSharedMemorySize, LDS_BYTES);

    pack_x_kernel<<<(B_ * HP * 32) / 256, 256, 0, stream>>>(x, wx);
    pack_w_kernel<<<(9 * OC_ * C_ + 255) / 256, 256, 0, stream>>>(w, ww);
    // 8192 waves (2 M-tiles each), 8 waves/block -> 1024 blocks
    conv_wmma_kernel<<<(B_ * H_ * (W_ / 32)) / 8, 256, LDS_BYTES, stream>>>(wx, ww, out);
}
